// mMultiHeadSelfAttention_60172491817080
// MI455X (gfx1250) — matrix operand order repaired and checked by reading, not compiled, not run on hardware
//
#include <hip/hip_runtime.h>
#include <hip/hip_bf16.h>

#define B_  4
#define S_  1024
#define F_  4
#define D_  1024
#define H_  16
#define HD_ 64
#define EPS_ 1e-5f

typedef __attribute__((ext_vector_type(16))) _Float16 v16h;
typedef __attribute__((ext_vector_type(8)))  _Float16 v8h;
typedef __attribute__((ext_vector_type(8)))  float    v8f;

union V16HU { v16h v; v8h h[2]; };

// ---------------------------------------------------------------- DPP utils
constexpr int dpp8sel(int a, int b, int c, int d, int e, int f, int g, int h) {
  return a | (b << 3) | (c << 6) | (d << 9) | (e << 12) | (f << 15) |
         (g << 18) | (h << 21);
}

template <int SEL>
__device__ __forceinline__ float dpp8f(float x) {
  return __int_as_float(__builtin_amdgcn_mov_dpp8(__float_as_int(x), SEL));
}

// DPP16 row_ror:8 — rotate each 16-lane row by 8 (VALU, no LDS)
__device__ __forceinline__ float row_ror8f(float x) {
  const int i = __float_as_int(x);
  return __int_as_float(__builtin_amdgcn_update_dpp(i, i, 0x128, 0xf, 0xf, true));
}

// max over each 16-lane group, entirely in VALU (dpp8 xor1/2/4 + dpp16 ror8)
__device__ __forceinline__ float red_max16(float v) {
  v = fmaxf(v, dpp8f<dpp8sel(1, 0, 3, 2, 5, 4, 7, 6)>(v));   // xor 1
  v = fmaxf(v, dpp8f<dpp8sel(2, 3, 0, 1, 6, 7, 4, 5)>(v));   // xor 2
  v = fmaxf(v, dpp8f<dpp8sel(4, 5, 6, 7, 0, 1, 2, 3)>(v));   // xor 4
  v = fmaxf(v, row_ror8f(v));                                // swap 8-groups
  return v;
}

// A-fragment (16x32 f16, row-major source, row stride ld):
// lanes 0-15 rows 0-15, halves 0..7 = K[hf..hf+7], halves 8..15 = K[16+hf..23+hf]
__device__ __forceinline__ v16h load_frag_a(const _Float16* base, int ld, int lane) {
  const int m  = lane & 15;
  const int hf = (lane >> 4) << 3;                 // 0 or 8
  const _Float16* p = base + m * ld;
  V16HU u;
  u.h[0] = *(const v8h*)(p + hf);
  u.h[1] = *(const v8h*)(p + 16 + hf);
  return u.v;
}

// B-fragment (32x16 f16) from B^T storage (Bt[n][k], row stride ld):
// lane n = column; halves 0..7 = K[hf..hf+7], halves 8..15 = K[16+hf..23+hf]
__device__ __forceinline__ v16h load_frag_b(const _Float16* bt, int ld, int lane) {
  const int n  = lane & 15;
  const int kh = (lane >> 4) << 3;                 // 0 or 8
  const _Float16* p = bt + n * ld + kh;
  V16HU u;
  u.h[0] = *(const v8h*)(p);
  u.h[1] = *(const v8h*)(p + 16);
  return u.v;
}

__device__ __forceinline__ v8f wmma16(v16h a, v16h b, v8f c) {
  return __builtin_amdgcn_wmma_f32_16x16x32_f16(false, a, false, b, (short)0, c,
                                                false, false);
}

// ---------------------------------------------------------------- QKV proj
// q/k/v[b,s,d] = sum_{f<4} x[b,s,f] * W[f,d] + bias[d];  K-dim of 4 -> VALU.
// Q,K stored f16 [B,H,S,64] row-major; V stored f16 transposed [B,H,64,S].
__global__ void __launch_bounds__(256)
qkv_kernel(const float* __restrict__ x,
           const float* __restrict__ Wq, const float* __restrict__ bq,
           const float* __restrict__ Wk, const float* __restrict__ bk,
           const float* __restrict__ Wv, const float* __restrict__ bv,
           _Float16* __restrict__ Qh, _Float16* __restrict__ Kh,
           _Float16* __restrict__ Vt) {
  const int tid = blockIdx.x * 256 + threadIdx.x;        // B*S*D = 2^22
  const int d = tid & (D_ - 1);
  const int s = (tid >> 10) & (S_ - 1);
  const int b = tid >> 20;
  const float* xr = x + ((size_t)b * S_ + s) * F_;
  const float x0 = xr[0], x1 = xr[1], x2 = xr[2], x3 = xr[3];
  const float q = x0 * Wq[d] + x1 * Wq[D_ + d] + x2 * Wq[2 * D_ + d] + x3 * Wq[3 * D_ + d] + bq[d];
  const float k = x0 * Wk[d] + x1 * Wk[D_ + d] + x2 * Wk[2 * D_ + d] + x3 * Wk[3 * D_ + d] + bk[d];
  const float v = x0 * Wv[d] + x1 * Wv[D_ + d] + x2 * Wv[2 * D_ + d] + x3 * Wv[3 * D_ + d] + bv[d];
  const int h = d >> 6, hd = d & 63;
  const size_t bh = (size_t)b * H_ + h;
  Qh[(bh * S_ + s) * HD_ + hd] = (_Float16)q;
  Kh[(bh * S_ + s) * HD_ + hd] = (_Float16)k;
  Vt[(bh * HD_ + hd) * S_ + s] = (_Float16)v;
}

// ------------------------------------------------------- Wo transpose->f16
__global__ void __launch_bounds__(256)
wot_kernel(const float* __restrict__ Wo, _Float16* __restrict__ WoT) {
  const int tid  = blockIdx.x * 256 + threadIdx.x;       // D*D
  const int dout = tid & (D_ - 1);
  const int din  = tid >> 10;
  WoT[(size_t)dout * D_ + din] = (_Float16)Wo[tid];
}

// ---------------------------------------------------- flash attention, WMMA
// One wave per (b, h, 16-row Q tile). Online softmax over 32-key steps.
// Denominator folded into a 5th WMMA accumulator (l = alpha*l + P@ones);
// row-max reduction is pure VALU DPP (no LDS round-trips).
__global__ void __launch_bounds__(32)
attn_kernel(const _Float16* __restrict__ Qh, const _Float16* __restrict__ Kh,
            const _Float16* __restrict__ Vt, _Float16* __restrict__ AOh) {
  __shared__ __align__(16) _Float16 Pl[16 * 32];
  const int lane = threadIdx.x & 31;
  const int qt = blockIdx.x & 63;                        // S/16 tiles
  const int bh = blockIdx.x >> 6;                        // b*H + h
  const int b  = bh >> 4;
  const int h  = bh & 15;

  const _Float16* Qb = Qh + (size_t)bh * S_ * HD_;
  const _Float16* Kb = Kh + (size_t)bh * S_ * HD_;
  const _Float16* Vb = Vt + (size_t)bh * HD_ * S_;

  const v16h a0 = load_frag_a(Qb + qt * 16 * HD_,      HD_, lane);  // d 0..31
  const v16h a1 = load_frag_a(Qb + qt * 16 * HD_ + 32, HD_, lane);  // d 32..63

  v16h ones;
#pragma unroll
  for (int i = 0; i < 16; ++i) ones[i] = (_Float16)1.0f;

  v8f o0 = {}, o1 = {}, o2 = {}, o3 = {};                // d chunks of 16
  v8f lacc = {};                                         // softmax denominators
  float m[8];
#pragma unroll
  for (int r = 0; r < 8; ++r) m[r] = -3.0e38f;

  const int col    = lane & 15;
  const int rowoff = (lane >> 4) << 3;                   // C-layout: row = r + rowoff

  for (int k0 = 0; k0 < S_; k0 += 32) {
    // scores: B = K^T (d x key); Bt = K row-major
    v16h bk00 = load_frag_b(Kb + (size_t)(k0     ) * HD_,      HD_, lane);
    v16h bk01 = load_frag_b(Kb + (size_t)(k0     ) * HD_ + 32, HD_, lane);
    v16h bk10 = load_frag_b(Kb + (size_t)(k0 + 16) * HD_,      HD_, lane);
    v16h bk11 = load_frag_b(Kb + (size_t)(k0 + 16) * HD_ + 32, HD_, lane);

    v8f s0 = {}; s0 = wmma16(a0, bk00, s0); s0 = wmma16(a1, bk01, s0);
    v8f s1 = {}; s1 = wmma16(a0, bk10, s1); s1 = wmma16(a1, bk11, s1);

    // per-row online softmax; P goes straight to LDS (C-layout -> A-layout)
#pragma unroll
    for (int r = 0; r < 8; ++r) {
      const float t0 = s0[r] * 0.125f;                   // 1/sqrt(64)
      const float t1 = s1[r] * 0.125f;
      const float rmax = red_max16(fmaxf(t0, t1));
      const float mn = fmaxf(m[r], rmax);
      const float alpha = __expf(m[r] - mn);
      m[r] = mn;
      o0[r] *= alpha; o1[r] *= alpha;
      o2[r] *= alpha; o3[r] *= alpha;
      lacc[r] *= alpha;
      const int row = r + rowoff;
      Pl[row * 32 + col]      = (_Float16)__expf(t0 - mn);
      Pl[row * 32 + 16 + col] = (_Float16)__expf(t1 - mn);
    }
    __syncthreads();
    const v16h pa = load_frag_a(Pl, 32, lane);
    // O += P (16x32) @ V (32x64); B = V (key x d); Bt = Vt (d x key)
    o0 = wmma16(pa, load_frag_b(Vb + (size_t) 0 * S_ + k0, S_, lane), o0);
    o1 = wmma16(pa, load_frag_b(Vb + (size_t)16 * S_ + k0, S_, lane), o1);
    o2 = wmma16(pa, load_frag_b(Vb + (size_t)32 * S_ + k0, S_, lane), o2);
    o3 = wmma16(pa, load_frag_b(Vb + (size_t)48 * S_ + k0, S_, lane), o3);
    lacc = wmma16(pa, ones, lacc);                       // l += rowsum(P)
    __syncthreads();
  }

  _Float16* outp = AOh + (size_t)(b * S_ + qt * 16) * D_ + h * HD_;
#pragma unroll
  for (int r = 0; r < 8; ++r) {
    const int row = r + rowoff;
    const float inv = 1.0f / lacc[r];
    outp[(size_t)row * D_ +  0 + col] = (_Float16)(o0[r] * inv);
    outp[(size_t)row * D_ + 16 + col] = (_Float16)(o1[r] * inv);
    outp[(size_t)row * D_ + 32 + col] = (_Float16)(o2[r] * inv);
    outp[(size_t)row * D_ + 48 + col] = (_Float16)(o3[r] * inv);
  }
}

// ------------------------------------------------ out @ Wo + bo (WMMA GEMM)
// Each wave computes a 16x64 tile: 4 WMMAs per A-fragment load.
__global__ void __launch_bounds__(128)
wo_gemm_kernel(const _Float16* __restrict__ AOh, const _Float16* __restrict__ WoT,
               const float* __restrict__ bo, float* __restrict__ out2) {
  const int lane = threadIdx.x & 31;
  const int wave = threadIdx.x >> 5;
  const int tile = blockIdx.x * 4 + wave;                // 256 * 16 tiles of 16x64
  const int mt = tile >> 4;
  const int nb = tile & 15;                              // 64-wide N block
  const _Float16* Abase = AOh + (size_t)mt * 16 * D_;
  const _Float16* Bbase = WoT + (size_t)nb * 64 * D_;
  v8f c0 = {}, c1 = {}, c2 = {}, c3 = {};
  for (int k0 = 0; k0 < D_; k0 += 32) {
    __builtin_prefetch(Abase + k0 + 256, 0, 1);          // global_prefetch_b8
    const v16h a = load_frag_a(Abase + k0, D_, lane);
    c0 = wmma16(a, load_frag_b(Bbase + (size_t) 0 * D_ + k0, D_, lane), c0);
    c1 = wmma16(a, load_frag_b(Bbase + (size_t)16 * D_ + k0, D_, lane), c1);
    c2 = wmma16(a, load_frag_b(Bbase + (size_t)32 * D_ + k0, D_, lane), c2);
    c3 = wmma16(a, load_frag_b(Bbase + (size_t)48 * D_ + k0, D_, lane), c3);
  }
  const int col    = lane & 15;
  const int rowoff = (lane >> 4) << 3;
  float* orow = out2 + (size_t)(mt * 16) * D_ + nb * 64 + col;
  const float bias0 = bo[nb * 64 +  0 + col];
  const float bias1 = bo[nb * 64 + 16 + col];
  const float bias2 = bo[nb * 64 + 32 + col];
  const float bias3 = bo[nb * 64 + 48 + col];
#pragma unroll
  for (int r = 0; r < 8; ++r) {
    const size_t ro = (size_t)(r + rowoff) * D_;
    orow[ro +  0] = c0[r] + bias0;
    orow[ro + 16] = c1[r] + bias1;
    orow[ro + 32] = c2[r] + bias2;
    orow[ro + 48] = c3[r] + bias3;
  }
}

// --------------------------------------- conv1 (D->4 over channels) + ReLU
__global__ void __launch_bounds__(256)
conv1_kernel(const float* __restrict__ out2, const float* __restrict__ w,
             const float* __restrict__ bias, float* __restrict__ r1) {
  __shared__ float sh[4 * 256];
  const int t  = threadIdx.x;
  const int bs = blockIdx.x;                             // b*S + s
  const int b  = bs >> 10;
  const int s  = bs & 1023;
  const float* row = out2 + (size_t)bs * D_;
  float a0 = 0, a1 = 0, a2 = 0, a3 = 0;
  for (int d = t; d < D_; d += 256) {
    const float v = row[d];
    a0 += w[0 * D_ + d] * v;
    a1 += w[1 * D_ + d] * v;
    a2 += w[2 * D_ + d] * v;
    a3 += w[3 * D_ + d] * v;
  }
  sh[0 * 256 + t] = a0; sh[1 * 256 + t] = a1;
  sh[2 * 256 + t] = a2; sh[3 * 256 + t] = a3;
  __syncthreads();
  for (int off = 128; off > 0; off >>= 1) {
    if (t < off) {
#pragma unroll
      for (int o = 0; o < 4; ++o) sh[o * 256 + t] += sh[o * 256 + t + off];
    }
    __syncthreads();
  }
  if (t < 4)
    r1[((size_t)b * 4 + t) * S_ + s] = fmaxf(sh[t * 256] + bias[t], 0.f);
}

// ----------------------------------------- conv2 (4S->S) + BatchNorm + ReLU
__global__ void __launch_bounds__(256)
conv2_kernel(const float* __restrict__ r1, const float* __restrict__ w2,
             const float* __restrict__ b2, const float* __restrict__ gamma,
             const float* __restrict__ beta, const float* __restrict__ mean,
             const float* __restrict__ var, float* __restrict__ t2) {
  __shared__ float sh[256];
  const int t  = threadIdx.x;
  const int bj = blockIdx.x;                             // b*S + j
  const int b  = bj >> 10;
  const int j  = bj & 1023;
  const float* rb = r1 + (size_t)b * 4 * S_;
  const float* wr = w2 + (size_t)j * 4 * S_;
  float acc = 0;
  for (int i = t; i < 4 * S_; i += 256) acc += rb[i] * wr[i];
  sh[t] = acc; __syncthreads();
  for (int off = 128; off > 0; off >>= 1) {
    if (t < off) sh[t] += sh[t + off];
    __syncthreads();
  }
  if (t == 0) {
    float y = sh[0] + b2[j];
    y = (y - mean[j]) * rsqrtf(var[j] + EPS_) * gamma[j] + beta[j];
    t2[(size_t)b * S_ + j] = fmaxf(y, 0.f);
  }
}

// ------------------------------------------------------------ conv3 (S->2)
__global__ void __launch_bounds__(256)
conv3_kernel(const float* __restrict__ t2, const float* __restrict__ w3,
             const float* __restrict__ b3, float* __restrict__ out) {
  __shared__ float sh[256];
  const int t  = threadIdx.x;
  const int b  = blockIdx.x >> 1;
  const int cc = blockIdx.x & 1;
  const float* tb = t2 + (size_t)b * S_;
  const float* wr = w3 + (size_t)cc * S_;
  float acc = 0;
  for (int j = t; j < S_; j += 256) acc += tb[j] * wr[j];
  sh[t] = acc; __syncthreads();
  for (int off = 128; off > 0; off >>= 1) {
    if (t < off) sh[t] += sh[t + off];
    __syncthreads();
  }
  if (t == 0) out[b * 2 + cc] = sh[0] + b3[cc];
}

extern "C" void kernel_launch(void* const* d_in, const int* in_sizes, int n_in,
                              void* d_out, int out_size, void* d_ws, size_t ws_size,
                              hipStream_t stream) {
  const float* x   = (const float*)d_in[0];
  const float* Wq  = (const float*)d_in[1];
  const float* bq  = (const float*)d_in[2];
  const float* Wk  = (const float*)d_in[3];
  const float* bk  = (const float*)d_in[4];
  const float* Wv  = (const float*)d_in[5];
  const float* bv  = (const float*)d_in[6];
  const float* Wo  = (const float*)d_in[7];
  const float* bo  = (const float*)d_in[8];
  const float* c1w = (const float*)d_in[9];
  const float* c1b = (const float*)d_in[10];
  const float* c2w = (const float*)d_in[11];
  const float* c2b = (const float*)d_in[12];
  const float* g   = (const float*)d_in[13];
  const float* bta = (const float*)d_in[14];
  const float* mu  = (const float*)d_in[15];
  const float* var = (const float*)d_in[16];
  const float* c3w = (const float*)d_in[17];
  const float* c3b = (const float*)d_in[18];
  float* out = (float*)d_out;
  (void)in_sizes; (void)n_in; (void)out_size; (void)ws_size;

  char* ws = (char*)d_ws;
  size_t off = 0;
  auto take = [&](size_t bytes) -> char* {
    char* p = ws + off;
    off += (bytes + 255) & ~(size_t)255;
    return p;
  };
  _Float16* Qh  = (_Float16*)take((size_t)B_ * H_ * S_ * HD_ * 2);   // 8 MB
  _Float16* Kh  = (_Float16*)take((size_t)B_ * H_ * S_ * HD_ * 2);   // 8 MB
  _Float16* Vt  = (_Float16*)take((size_t)B_ * H_ * HD_ * S_ * 2);   // 8 MB
  _Float16* AOh = (_Float16*)take((size_t)B_ * S_ * D_ * 2);         // 8 MB
  _Float16* WoT = (_Float16*)take((size_t)D_ * D_ * 2);              // 2 MB
  float*    out2 = (float*)take((size_t)B_ * S_ * D_ * 4);           // 16 MB
  float*    r1   = (float*)take((size_t)B_ * 4 * S_ * 4);
  float*    t2   = (float*)take((size_t)B_ * S_ * 4);

  qkv_kernel<<<(B_ * S_ * D_) / 256, 256, 0, stream>>>(x, Wq, bq, Wk, bk, Wv, bv,
                                                       Qh, Kh, Vt);
  wot_kernel<<<(D_ * D_) / 256, 256, 0, stream>>>(Wo, WoT);
  attn_kernel<<<B_ * H_ * (S_ / 16), 32, 0, stream>>>(Qh, Kh, Vt, AOh);
  wo_gemm_kernel<<<(B_ * S_ / 16) * (D_ / 64) / 4, 128, 0, stream>>>(AOh, WoT, bo, out2);
  conv1_kernel<<<B_ * S_, 256, 0, stream>>>(out2, c1w, c1b, r1);
  conv2_kernel<<<B_ * S_, 256, 0, stream>>>(r1, c2w, c2b, g, bta, mu, var, t2);
  conv3_kernel<<<B_ * 2, 256, 0, stream>>>(t2, c3w, c3b, out);
}
